// GCNModel_58901181498010
// MI455X (gfx1250) — matrix operand order repaired and checked by reading, not compiled, not run on hardware
//
#include <hip/hip_runtime.h>

// ---------------------------------------------------------------------------
// GCN (2x GCNConv + FC) for MI455X / gfx1250, wave32.
//  - GEMMs: v_wmma_f32_16x16x32_bf16, A tiles staged with double-buffered
//    global_load_async_to_lds_b128 (ASYNCcnt), W parked in LDS frag layout.
//  - Self-loop init fused into GEMM epilogue (bufB = dinv^2 * h).
//  - Edge norm precomputed once (shared by both layers), indices packed int32.
//  - Scatter-add via hardware global_atomic_add_f32 (L2-resident working set).
// ---------------------------------------------------------------------------

typedef __attribute__((ext_vector_type(16))) __bf16          v16bf;
typedef __attribute__((ext_vector_type(8)))  float           v8f;
typedef __attribute__((ext_vector_type(8)))  unsigned short  u16x8;
typedef __attribute__((ext_vector_type(16))) unsigned short  u16x16;

#define IN_DIM 512
#define HID    64
#define OUTD   5

__device__ __forceinline__ unsigned short f2bf(float f) {
  // round-to-nearest-even fp32 -> bf16
  unsigned int u = __builtin_bit_cast(unsigned int, f);
  unsigned int r = u + 0x7FFFu + ((u >> 16) & 1u);
  return (unsigned short)(r >> 16);
}

__device__ __forceinline__ void async_ld_b128(unsigned lds_off, const float* gaddr) {
  // CDNA5 async copy: global -> LDS, tracked by ASYNCcnt, no VGPR transit.
  asm volatile("global_load_async_to_lds_b128 %0, %1, off"
               :: "v"(lds_off), "v"(gaddr) : "memory");
}
__device__ __forceinline__ void wait_async0() {
  asm volatile("s_wait_asynccnt 0" ::: "memory");
}

// ------------------------------ degree / dinv ------------------------------

__global__ void k_deg_init(float* __restrict__ deg, int n) {
  int i = blockIdx.x * blockDim.x + threadIdx.x;
  if (i < n) deg[i] = 1.0f;   // self-loop weight
}

__global__ void k_deg_edges(const long long* __restrict__ dst,
                            const float* __restrict__ ew,
                            float* __restrict__ deg, int E) {
  int e = blockIdx.x * blockDim.x + threadIdx.x;
  if (e < E) unsafeAtomicAdd(&deg[(int)dst[e]], ew[e]);
}

__global__ void k_dinv(float* __restrict__ deg, int n) {
  int i = blockIdx.x * blockDim.x + threadIdx.x;
  if (i < n) {
    float d = deg[i];
    deg[i] = d > 0.0f ? rsqrtf(d) : 0.0f;
  }
}

// Pack indices to int32 and precompute norm = dinv[s]*ew*dinv[d] (graph-only,
// reused by both layers).
__global__ void k_prep_edges(const long long* __restrict__ src,
                             const long long* __restrict__ dst,
                             const float* __restrict__ ew,
                             const float* __restrict__ dinv,
                             int2* __restrict__ idx2,
                             float* __restrict__ nrm, int E) {
  int e = blockIdx.x * blockDim.x + threadIdx.x;
  if (e < E) {
    int s = (int)src[e];
    int d = (int)dst[e];
    idx2[e] = make_int2(s, d);
    nrm[e]  = dinv[s] * ew[e] * dinv[d];
  }
}

// ------------------------------ WMMA GEMM ----------------------------------
// Y[M][64] = X[M][K] @ W[K][64]  (bf16 WMMA, fp32 acc)
// Yl[r][:] = dinv[r]^2 * Y[r][:] (fused self-loop init of the aggregation buf)
// Block: 256 threads = 8 waves; wave owns 16 rows x 64 cols (4 C tiles).
// A tiles: fp32, double-buffered in LDS via global_load_async_to_lds_b128.

template <int K>
__global__ __launch_bounds__(256) void k_gemm_bf16(const float* __restrict__ X,
                                                   const float* __restrict__ W,
                                                   float* __restrict__ Y,
                                                   float* __restrict__ Yl,
                                                   const float* __restrict__ dinv,
                                                   int M) {
  constexpr int KC = K / 32;                       // k-slabs of 32
  __shared__ __align__(32) unsigned short sW[KC * 4 * 32 * 16]; // [c][nt][lane][i]
  __shared__ __align__(32) float sAf[2][128 * 36]; // 128 rows x 32 f32, stride 36

  const int tid  = threadIdx.x;
  const int lane = tid & 31;
  const int wave = tid >> 5;
  const int rowBase = blockIdx.x * 128;

  // Stage W (K x 64) into bf16 B-fragment layout: K = (i&7) + 8*(lane>=16) + 16*(i>=8), N = lane%16.
  for (int t = tid; t < K * 64; t += 256) {
    int i  = t & 15;
    int l  = (t >> 4) & 31;
    int nt = (t >> 9) & 3;
    int c  = t >> 11;
    int k  = c * 32 + (i & 7) + ((l >> 4) << 3) + ((i >> 3) << 4);
    int n  = nt * 16 + (l & 15);
    sW[t] = f2bf(W[k * 64 + n]);
  }

  // Async fill of one 128x32 fp32 slab into LDS buffer b (4 x b128 per thread).
  auto fill = [&](int c, int b) {
#pragma unroll
    for (int j = 0; j < 4; ++j) {
      int q  = tid + j * 256;
      int r  = q >> 3;
      int c4 = q & 7;
      int rg = rowBase + r; if (rg > M - 1) rg = M - 1;
      unsigned lofs = (unsigned)(size_t)(&sAf[b][r * 36 + c4 * 4]);
      async_ld_b128(lofs, X + (size_t)rg * K + c * 32 + c4 * 4);
    }
  };

  v8f acc[4];
#pragma unroll
  for (int nt = 0; nt < 4; ++nt) acc[nt] = (v8f){0, 0, 0, 0, 0, 0, 0, 0};

  const int myRowL = wave * 16 + (lane & 15);   // A frag: M = lane%16
  const int koff   = (lane >> 4) * 8;           // fp32 elems: 0 or 8

  fill(0, 0);
  wait_async0();
  __syncthreads();   // tile 0 + staged W visible to all waves

#pragma unroll 1
  for (int c = 0; c < KC; ++c) {
    if (c + 1 < KC) fill(c + 1, (c + 1) & 1);   // prefetch next slab (async)

    // A fragment: 4x float4 from LDS, convert to bf16 at read time.
    const float* ap = &sAf[c & 1][myRowL * 36 + koff];
    float4 p0 = *(const float4*)(ap);
    float4 p1 = *(const float4*)(ap + 4);
    float4 p2 = *(const float4*)(ap + 16);
    float4 p3 = *(const float4*)(ap + 20);
    u16x16 av = (u16x16){
        f2bf(p0.x), f2bf(p0.y), f2bf(p0.z), f2bf(p0.w),
        f2bf(p1.x), f2bf(p1.y), f2bf(p1.z), f2bf(p1.w),
        f2bf(p2.x), f2bf(p2.y), f2bf(p2.z), f2bf(p2.w),
        f2bf(p3.x), f2bf(p3.y), f2bf(p3.z), f2bf(p3.w)};
    v16bf af = __builtin_bit_cast(v16bf, av);

#pragma unroll
    for (int nt = 0; nt < 4; ++nt) {
      const unsigned short* bp = &sW[(((c * 4 + nt) * 32) + lane) * 16];
      u16x8 b0 = *(const u16x8*)bp;
      u16x8 b1 = *(const u16x8*)(bp + 8);
      u16x16 bv = __builtin_shufflevector(b0, b1, 0, 1, 2, 3, 4, 5, 6, 7,
                                          8, 9, 10, 11, 12, 13, 14, 15);
      v16bf bf = __builtin_bit_cast(v16bf, bv);
      acc[nt] = __builtin_amdgcn_wmma_f32_16x16x32_bf16(
          false, af, false, bf, (short)0, acc[nt], false, false);
    }

    wait_async0();     // next slab landed (issued before compute -> overlapped)
    __syncthreads();
  }

  // Epilogue: C/D layout M = v + 8*(lane>=16), N = lane%16; coalesced per half-wave.
  // Writes h (Y) and the fused self-loop-initialized aggregation buffer (Yl).
  const int col  = lane & 15;
  const int rOff = (lane >> 4) * 8;
#pragma unroll
  for (int v = 0; v < 8; ++v) {
    int r  = rowBase + wave * 16 + rOff + v;
    int rc = r < M ? r : M - 1;
    float s = dinv[rc];
    s = s * s;
    if (r < M) {
#pragma unroll
      for (int nt = 0; nt < 4; ++nt) {
        float val = acc[nt][v];
        Y [(size_t)r * 64 + nt * 16 + col] = val;
        Yl[(size_t)r * 64 + nt * 16 + col] = s * val;
      }
    }
  }
}

// --------------------------- edge aggregation ------------------------------

// out[dst][:] += norm * h[src][:]  -- 2 lanes per edge, 32 cols each.
__global__ void k_scatter(const int2* __restrict__ idx2,
                          const float* __restrict__ nrm,
                          const float* __restrict__ h,
                          float* __restrict__ out, int E) {
  int t = blockIdx.x * blockDim.x + threadIdx.x;
  if (t >= E * 2) return;
  int e    = t >> 1;
  int half = t & 1;
  int2 sd  = idx2[e];
  float n  = nrm[e];
  const float4* hp = (const float4*)(h + (size_t)sd.x * 64 + half * 32);
  float* op = out + (size_t)sd.y * 64 + half * 32;
#pragma unroll
  for (int j = 0; j < 8; ++j) {
    float4 v = hp[j];
    unsafeAtomicAdd(op + j * 4 + 0, n * v.x);
    unsafeAtomicAdd(op + j * 4 + 1, n * v.y);
    unsafeAtomicAdd(op + j * 4 + 2, n * v.z);
    unsafeAtomicAdd(op + j * 4 + 3, n * v.w);
  }
}

__global__ void k_bias_relu(float* __restrict__ h, const float* __restrict__ b, int n) {
  int t = blockIdx.x * blockDim.x + threadIdx.x;
  if (t < n * 64) {
    float v = h[t] + b[t & 63];
    h[t] = v > 0.0f ? v : 0.0f;
  }
}

// ------------------------------- FC head -----------------------------------
// out = relu(h + b2) @ Wfc + bfc   (layer-2 bias+ReLU fused in)

__global__ void k_fc(const float* __restrict__ h, const float* __restrict__ Wfc,
                     const float* __restrict__ bfc, const float* __restrict__ b2,
                     float* __restrict__ out, int n) {
  __shared__ float sW[HID * OUTD];
  __shared__ float sb[OUTD];
  __shared__ float sb2[HID];
  int tid = threadIdx.x;
  if (tid < HID * OUTD) sW[tid] = Wfc[tid];
  if (tid < OUTD) sb[tid] = bfc[tid];
  if (tid < HID) sb2[tid] = b2[tid];
  __syncthreads();
  int i = blockIdx.x * blockDim.x + tid;
  if (i < n) {
    float a0 = sb[0], a1 = sb[1], a2 = sb[2], a3 = sb[3], a4 = sb[4];
    const float* hp = h + (size_t)i * HID;
#pragma unroll
    for (int j = 0; j < HID; ++j) {
      float x = hp[j] + sb2[j];
      x = x > 0.0f ? x : 0.0f;
      a0 += x * sW[j * OUTD + 0];
      a1 += x * sW[j * OUTD + 1];
      a2 += x * sW[j * OUTD + 2];
      a3 += x * sW[j * OUTD + 3];
      a4 += x * sW[j * OUTD + 4];
    }
    float* op = out + (size_t)i * OUTD;
    op[0] = a0; op[1] = a1; op[2] = a2; op[3] = a3; op[4] = a4;
  }
}

// ------------------------------- launcher ----------------------------------

extern "C" void kernel_launch(void* const* d_in, const int* in_sizes, int n_in,
                              void* d_out, int out_size, void* d_ws, size_t ws_size,
                              hipStream_t stream) {
  const float*     x   = (const float*)d_in[0];
  const long long* ei  = (const long long*)d_in[1];
  const float*     ew  = (const float*)d_in[2];
  const float*     W1  = (const float*)d_in[3];
  const float*     b1  = (const float*)d_in[4];
  const float*     W2  = (const float*)d_in[5];
  const float*     b2  = (const float*)d_in[6];
  const float*     Wfc = (const float*)d_in[7];
  const float*     bfc = (const float*)d_in[8];

  const int NN = in_sizes[0] / IN_DIM;
  const int E  = in_sizes[2];
  const long long* src = ei;       // edge_index[0]
  const long long* dst = ei + E;   // edge_index[1]

  // workspace layout
  char* ws = (char*)d_ws;
  auto align256 = [](size_t v) { return (v + 255) & ~(size_t)255; };
  size_t o = 0;
  float* dinv = (float*)(ws + o);  o += align256((size_t)NN * 4);
  float* bufA = (float*)(ws + o);  o += align256((size_t)NN * HID * 4);
  float* bufB = (float*)(ws + o);  o += align256((size_t)NN * HID * 4);
  int2*  idx2 = (int2*) (ws + o);  o += align256((size_t)E * 8);
  float* nrm  = (float*)(ws + o);  o += align256((size_t)E * 4);

  const int B = 256;
  const int gmNN = (NN + B - 1) / B;
  const int gmE  = (E + B - 1) / B;
  const int gmE2 = (E * 2 + B - 1) / B;
  const int gm64 = (NN * 64 + B - 1) / B;
  const int gmM  = (NN + 127) / 128;

  // graph normalization (computed once, reused by both layers)
  k_deg_init<<<gmNN, B, 0, stream>>>(dinv, NN);
  k_deg_edges<<<gmE, B, 0, stream>>>(dst, ew, dinv, E);
  k_dinv<<<gmNN, B, 0, stream>>>(dinv, NN);
  k_prep_edges<<<gmE, B, 0, stream>>>(src, dst, ew, dinv, idx2, nrm, E);

  // layer 1: bufA = x@W1 ; bufB = dinv^2*bufA (fused) ; scatter ; bias+relu
  k_gemm_bf16<IN_DIM><<<gmM, B, 0, stream>>>(x, W1, bufA, bufB, dinv, NN);
  k_scatter<<<gmE2, B, 0, stream>>>(idx2, nrm, bufA, bufB, E);
  k_bias_relu<<<gm64, B, 0, stream>>>(bufB, b1, NN);

  // layer 2: bufA = bufB@W2 ; bufB = dinv^2*bufA in place (block-local rows) ;
  // scatter ; bias+relu fused into FC head
  k_gemm_bf16<HID><<<gmM, B, 0, stream>>>(bufB, W2, bufA, bufB, dinv, NN);
  k_scatter<<<gmE2, B, 0, stream>>>(idx2, nrm, bufA, bufB, E);

  // head
  k_fc<<<gmNN, B, 0, stream>>>(bufB, Wfc, bfc, b2, (float*)d_out, NN);
}
